// CutieReadDecodeForOnnx_22806276342489
// MI455X (gfx1250) — matrix operand order repaired and checked by reading, not compiled, not run on hardware
//
#include <hip/hip_runtime.h>
#include <hip/hip_bf16.h>

// ---------------------------------------------------------------------------
// Cutie memory read for MI455X (gfx1250, wave32, WMMA).
//   sim GEMM (K=132, fp32)  -> V_WMMA_F32_16X16X4_F32
//   readout GEMM (K=25920)  -> V_WMMA_F32_16X16X32_BF16
// Softmax uses shift=0 (sim <= 0 provably, since qe,ms >= 0), unnormalized
// exp stored bf16 transposed [P,N]; normalization folded into GEMM epilogue.
// ---------------------------------------------------------------------------

typedef __bf16 bf16_t;
typedef __attribute__((ext_vector_type(16))) __bf16 v16bf;
typedef __attribute__((ext_vector_type(8)))  __bf16 v8bf;
typedef __attribute__((ext_vector_type(8)))  float  v8f;
typedef __attribute__((ext_vector_type(4)))  float  v4f;
typedef __attribute__((ext_vector_type(2)))  float  v2f;

#define KDIM 64
#define NMEM 25920   // 16*30*54 memory elements
#define PQ   1620    // 30*54 query pixels
#define VDIM 1536    // 3 objects * 512 value dim

// ------------------------------- utility -----------------------------------

__global__ __launch_bounds__(256) void k_zero_f32(float* __restrict__ p, int n) {
    int i = blockIdx.x * 256 + threadIdx.x;
    if (i < n) p[i] = 0.0f;
}

// value f32 [V,N] -> bf16 [V,N]
__global__ __launch_bounds__(256) void k_convert_value_bf16(
    const float* __restrict__ in, bf16_t* __restrict__ out, int n) {
    int stride = gridDim.x * 256 * 8;
    for (int i = (blockIdx.x * 256 + threadIdx.x) * 8; i < n; i += stride) {
        v4f a = *(const v4f*)(in + i);
        v4f b = *(const v4f*)(in + i + 4);
        v8bf o;
        o[0] = (bf16_t)a.x; o[1] = (bf16_t)a.y; o[2] = (bf16_t)a.z; o[3] = (bf16_t)a.w;
        o[4] = (bf16_t)b.x; o[5] = (bf16_t)b.y; o[6] = (bf16_t)b.z; o[7] = (bf16_t)b.w;
        *(v8bf*)(out + i) = o;
    }
}

// ------------------------- stage 1: sim + exp ------------------------------
// Block tile: 64 n x 32 p, 8 waves (4 along n, 2 along p), each wave one
// 16x16 tile via 33 steps of V_WMMA_F32_16X16X4_F32 (K = 132).
//   A2[k][n]: k<64: 2*mk ; 64<=k<128: -mk^2 ; k=128: -1 ; k>128: 0
//   B2[k][p]: k<64: qk*qe; 64<=k<128: qe    ; k=128: sum_c qe*qk^2 ; k>128: 0
// sim = (A2^T B2)[n,p] * ms[n] / 8 ; e = exp(sim) stored bf16 at affT[p*N+n].

__global__ __launch_bounds__(256) void k_sim_exp(
    const float* __restrict__ mk,   // [64, NMEM]
    const float* __restrict__ ms,   // [NMEM]
    const float* __restrict__ qk,   // [64, PQ]
    const float* __restrict__ qe,   // [64, PQ]
    bf16_t* __restrict__ affT,      // [PQ, NMEM]
    float* __restrict__ colSum)     // [PQ]
{
    __shared__ float A2s[132][64];
    __shared__ float B2s[132][32];
    __shared__ float qks[64][32];
    __shared__ float eS[64][33];    // +1 pad: conflict-free transpose
    __shared__ float msS[64];
    __shared__ float partial[32][8];

    const int tid = threadIdx.x;
    const int n0  = blockIdx.x * 64;
    const int p0  = blockIdx.y * 32;

    // ---- stage operands into LDS (coalesced along n / p) ----
    for (int idx = tid; idx < 64 * 64; idx += 256) {
        int c = idx >> 6, i = idx & 63;
        float m = mk[(size_t)c * NMEM + n0 + i];
        A2s[c][i]      = 2.0f * m;
        A2s[64 + c][i] = -m * m;
    }
    if (tid < 64) {
        A2s[128][tid] = -1.0f;
        A2s[129][tid] = 0.0f; A2s[130][tid] = 0.0f; A2s[131][tid] = 0.0f;
        msS[tid] = ms[n0 + tid] * 0.125f;           // / sqrt(Ck=64)
    }
    for (int idx = tid; idx < 64 * 32; idx += 256) {
        int c = idx >> 5, j = idx & 31;
        int p = p0 + j;
        float kq = 0.0f, eq = 0.0f;
        if (p < PQ) { kq = qk[(size_t)c * PQ + p]; eq = qe[(size_t)c * PQ + p]; }
        B2s[c][j]      = kq * eq;
        B2s[64 + c][j] = eq;
        qks[c][j]      = kq;
    }
    if (tid < 32) { B2s[129][tid] = 0.0f; B2s[130][tid] = 0.0f; B2s[131][tid] = 0.0f; }
    __syncthreads();
    if (tid < 32) {                                  // b_sq[p] = sum_c qe*qk^2
        float s = 0.0f;
        #pragma unroll
        for (int c = 0; c < 64; ++c) s += B2s[c][tid] * qks[c][tid];
        B2s[128][tid] = s;
    }
    __syncthreads();

    // ---- per-wave 16x16 fp32 WMMA accumulation ----
    const int wave = tid >> 5, lane = tid & 31;
    const int wn = wave & 3, wp = wave >> 1 & 0;     // placeholder, fixed below
    (void)wp;
    const int wpg = wave >> 2;                       // 0..1 (p group)
    const int laneHalf = lane >> 4, l15 = lane & 15;
    const int nLoc = wn * 16 + l15;                  // A: M = lane&15
    const int pLoc = wpg * 16 + l15;                 // B/D: N = lane&15

    v8f acc = {};
    #pragma unroll
    for (int s = 0; s < 33; ++s) {
        // A 16x4: lanes 0-15 hold K=0,1 ; lanes 16-31 hold K=2,3
        int kk = s * 4 + 2 * laneHalf;
        v2f a, b;
        a.x = A2s[kk][nLoc];     a.y = A2s[kk + 1][nLoc];
        b.x = B2s[kk][pLoc];     b.y = B2s[kk + 1][pLoc];
        acc = __builtin_amdgcn_wmma_f32_16x16x4_f32(
            false, a, false, b, (short)0, acc, false, false);
    }

    // ---- epilogue: shrinkage scale + exp, transpose via LDS ----
    #pragma unroll
    for (int r = 0; r < 8; ++r) {
        int nl = wn * 16 + r + 8 * laneHalf;         // D: M = r + 8*(lane>=16)
        float sim = acc[r] * msS[nl];                // sim <= 0 -> shift-free exp
        eS[nl][pLoc] = __expf(sim);
    }
    __syncthreads();

    // ---- coalesced bf16 stores along n + column-sum reduction ----
    const int j  = tid >> 3;                         // p-local 0..31
    const int nb = (tid & 7) * 8;                    // n chunk
    float s = 0.0f;
    v8bf ov;
    #pragma unroll
    for (int i = 0; i < 8; ++i) {
        float v = eS[nb + i][j];
        s += v;
        ov[i] = (bf16_t)v;
    }
    partial[j][tid & 7] = s;
    const int p = p0 + j;
    if (p < PQ)
        *(v8bf*)(affT + (size_t)p * NMEM + n0 + nb) = ov;
    __syncthreads();
    if ((tid & 7) == 0 && p < PQ) {
        float t = 0.0f;
        #pragma unroll
        for (int i = 0; i < 8; ++i) t += partial[j][i];
        atomicAdd(colSum + p, t);
    }
}

// ---------------------- stage 2: value readout GEMM ------------------------
// out[v,p] = (1/colSum[p]) * sum_n valB[v,n] * affT[p,n]
// Block: 256 threads = 8 waves as 4(M) x 2(P); wave owns 64 rows x 16 cols
// (4 accumulators share one B fragment). Fragments load directly from global:
// valB (80 MB) + affT (84 MB) both live in the 192 MB L2 across the grid.
//   A frag (16-bit 16x32): lane row = lane&15, chunks k+8*(lane>=16) and
//                          k+16+8*(lane>=16), each 8 contiguous bf16.
//   B frag (16-bit 32x16): lane col = lane&15, same two 8-bf16 chunks as A.

__global__ __launch_bounds__(256) void k_readout_gemm(
    const bf16_t* __restrict__ valB,   // [VDIM, NMEM]
    const bf16_t* __restrict__ affT,   // [PQ, NMEM]
    const float* __restrict__ colSum,  // [PQ]
    float* __restrict__ out)           // [VDIM, PQ]
{
    const int tid = threadIdx.x;
    const int wave = tid >> 5, lane = tid & 31;
    const int mw = wave & 3, nw = wave >> 2;
    const int laneHalf = lane >> 4, l15 = lane & 15;

    const int m0   = blockIdx.x * 256 + mw * 64;
    const int pCol = blockIdx.y * 32 + nw * 16 + l15;
    const int pc   = (pCol < PQ) ? pCol : (PQ - 1);   // clamp: safe loads

    const bf16_t* bRow  = affT + (size_t)pc * NMEM + 8 * laneHalf;
    const bf16_t* aBase = valB + (size_t)(m0 + l15) * NMEM + 8 * laneHalf;
    const size_t rowStride16 = (size_t)16 * NMEM;

    v8f acc0 = {}, acc1 = {}, acc2 = {}, acc3 = {};
    union { v16bf v; v8bf h[2]; } a;

    #pragma unroll 2
    for (int k = 0; k < NMEM; k += 32) {
        union { v16bf v; v8bf h[2]; } bu;
        bu.h[0] = *(const v8bf*)(bRow + k);
        bu.h[1] = *(const v8bf*)(bRow + k + 16);
        v16bf b = bu.v;

        a.h[0] = *(const v8bf*)(aBase + k);
        a.h[1] = *(const v8bf*)(aBase + k + 16);
        acc0 = __builtin_amdgcn_wmma_f32_16x16x32_bf16(
            false, a.v, false, b, (short)0, acc0, false, false);

        a.h[0] = *(const v8bf*)(aBase + rowStride16 + k);
        a.h[1] = *(const v8bf*)(aBase + rowStride16 + k + 16);
        acc1 = __builtin_amdgcn_wmma_f32_16x16x32_bf16(
            false, a.v, false, b, (short)0, acc1, false, false);

        a.h[0] = *(const v8bf*)(aBase + 2 * rowStride16 + k);
        a.h[1] = *(const v8bf*)(aBase + 2 * rowStride16 + k + 16);
        acc2 = __builtin_amdgcn_wmma_f32_16x16x32_bf16(
            false, a.v, false, b, (short)0, acc2, false, false);

        a.h[0] = *(const v8bf*)(aBase + 3 * rowStride16 + k);
        a.h[1] = *(const v8bf*)(aBase + 3 * rowStride16 + k + 16);
        acc3 = __builtin_amdgcn_wmma_f32_16x16x32_bf16(
            false, a.v, false, b, (short)0, acc3, false, false);
    }

    const float inv = 1.0f / colSum[pc];             // softmax normalization
    if (pCol < PQ) {
        #pragma unroll
        for (int r = 0; r < 8; ++r) {
            int mr = r + 8 * laneHalf;               // D: M = r + 8*(lane>=16)
            out[(size_t)(m0 +  0 + mr) * PQ + pCol] = acc0[r] * inv;
            out[(size_t)(m0 + 16 + mr) * PQ + pCol] = acc1[r] * inv;
            out[(size_t)(m0 + 32 + mr) * PQ + pCol] = acc2[r] * inv;
            out[(size_t)(m0 + 48 + mr) * PQ + pCol] = acc3[r] * inv;
        }
    }
}

// ------------------------------- launcher ----------------------------------

extern "C" void kernel_launch(void* const* d_in, const int* in_sizes, int n_in,
                              void* d_out, int out_size, void* d_ws, size_t ws_size,
                              hipStream_t stream) {
    (void)in_sizes; (void)n_in; (void)out_size; (void)ws_size;
    const float* mk  = (const float*)d_in[0];   // [1,64,25920]
    const float* ms  = (const float*)d_in[1];   // [1,1,25920]
    const float* qk  = (const float*)d_in[2];   // [1,64,30,54]
    const float* qe  = (const float*)d_in[3];   // [1,64,30,54]
    const float* val = (const float*)d_in[4];   // [1,3,512,25920]
    float* out = (float*)d_out;

    // workspace layout (all 256B aligned): valB 79.6MB | affT 84MB | colSum
    char* ws = (char*)d_ws;
    bf16_t* valB = (bf16_t*)ws;
    size_t off = ((size_t)VDIM * NMEM * sizeof(bf16_t) + 255) & ~(size_t)255;
    bf16_t* affT = (bf16_t*)(ws + off);
    off += ((size_t)PQ * NMEM * sizeof(bf16_t) + 255) & ~(size_t)255;
    float* colSum = (float*)(ws + off);

    k_zero_f32<<<(PQ + 255) / 256, 256, 0, stream>>>(colSum, PQ);
    k_convert_value_bf16<<<1024, 256, 0, stream>>>(val, valB, VDIM * NMEM);

    dim3 g1(NMEM / 64, (PQ + 31) / 32);              // 405 x 51
    k_sim_exp<<<g1, 256, 0, stream>>>(mk, ms, qk, qe, affT, colSum);

    dim3 g2(VDIM / 256, (PQ + 31) / 32);             // 6 x 51
    k_readout_gemm<<<g2, 256, 0, stream>>>(valB, affT, colSum, out);
}
